// DenoisingModelV4_15118284882179
// MI455X (gfx1250) — compile-verified, matrix operand order checked by reading, not run on hardware
//
#include <hip/hip_runtime.h>
#include <cstddef>

// ---- CDNA5 WMMA vector types (wave32) ----
typedef __attribute__((ext_vector_type(16))) __bf16 v16bf;
typedef __attribute__((ext_vector_type(8)))  __bf16 v8bf;
typedef __attribute__((ext_vector_type(8)))  float  v8f;

#define NB 16
#define NO 1024
#define NC 4
#define NF 256
#define NH 8
#define D2 128
#define ND 256
#define NIc 64
#define NJc 64

static __device__ __forceinline__ int iclamp(int v, int lo, int hi) {
    return v < lo ? lo : (v > hi ? hi : v);
}

static __device__ __forceinline__ v16bf cat8(v8bf lo, v8bf hi) {
    return __builtin_shufflevector(lo, hi, 0, 1, 2, 3, 4, 5, 6, 7,
                                           8, 9, 10, 11, 12, 13, 14, 15);
}

// -------------------------------------------------------------------------
// Kernel 0: repack w_feat (NF,H,D2) and w_out (H*D2, ND) f32 -> bf16 WMMA
// B-fragment layout. Fragment = [32 lanes][16 bf16] contiguous (32B/lane).
// Lane L: column n_local = L%16, half lh = L/16.
// Element e in [0,16): k_local = (e<8 ? 0 : 16) + 8*lh + (e&7).
// -------------------------------------------------------------------------
__global__ void pack_weights(const float* __restrict__ w_feat,
                             const float* __restrict__ w_out,
                             __bf16* __restrict__ wf_pk,
                             __bf16* __restrict__ wo_pk) {
    int tid  = blockIdx.x * blockDim.x + threadIdx.x; // 32768 threads total
    int lane = tid & 31;
    int frag = tid >> 5;
    int nl = lane & 15;
    int lh = lane >> 4;
    if (frag < 512) {                       // w_feat frags: frag = (h*8+nt)*8+kt
        int kt = frag & 7;
        int nt = (frag >> 3) & 7;
        int h  = frag >> 6;
        __bf16* dst = wf_pk + (size_t)frag * 512 + lane * 16;
#pragma unroll
        for (int e = 0; e < 16; ++e) {
            int k = kt * 32 + ((e < 8) ? 0 : 16) + 8 * lh + (e & 7);
            int d = nt * 16 + nl;
            dst[e] = (__bf16)w_feat[(k * NH + h) * D2 + d];
        }
    } else {                                // w_out frags: f2 = nt*32 + kt
        int f2 = frag - 512;
        int kt = f2 & 31;
        int nt = f2 >> 5;
        __bf16* dst = wo_pk + (size_t)f2 * 512 + lane * 16;
#pragma unroll
        for (int e = 0; e < 16; ++e) {
            int k = kt * 32 + ((e < 8) ? 0 : 16) + 8 * lh + (e & 7);
            int n = nt * 16 + nl;
            dst[e] = (__bf16)w_out[k * ND + n];
        }
    }
}

// -------------------------------------------------------------------------
// Kernel 1: projection + 4-tap gather + attention logits/softmax.
// One 256-thread block per (b,o). Writes feats (bf16) and aw (f32) to ws.
// -------------------------------------------------------------------------
__global__ __launch_bounds__(256) void gather_feats(
    const float* __restrict__ center,       // (NB,NO,3)
    const float* __restrict__ cam_posquat,  // (NB,NC,7)
    const float* __restrict__ intrinsic,    // (NB,NC,6)
    const float* __restrict__ img_feat,     // (NB,NC,NI,NJ,NF)
    const float* __restrict__ w_attn,       // (NF,NH)
    const float* __restrict__ b_attn,       // (NH)
    __bf16* __restrict__ feats_ws,          // (NB*NO*NC, NF) bf16
    float*  __restrict__ aw_ws)             // (NB*NO, NC, NH) f32
{
    int row = blockIdx.x;                   // b*NO + o
    int b = row >> 10;
    int tid = threadIdx.x;

    __shared__ float s_feats[NC][NF];
    __shared__ float s_w[NC][4];
    __shared__ int   s_idx[NC][4];
    __shared__ float s_aw[NC][NH];

    if (tid < NC) {
        int c = tid;
        const float* cp = cam_posquat + (size_t)(b * NC + c) * 7;
        const float* it = intrinsic   + (size_t)(b * NC + c) * 6;
        const float* ct = center      + (size_t)row * 3;
        float qx = cp[3], qy = cp[4], qz = cp[5], qw = cp[6];
        float qn = rsqrtf(qx * qx + qy * qy + qz * qz + qw * qw);
        qx *= qn; qy *= qn; qz *= qn; qw *= qn;
        float R00 = 1.f - 2.f * (qy * qy + qz * qz), R01 = 2.f * (qx * qy - qz * qw), R02 = 2.f * (qx * qz + qy * qw);
        float R10 = 2.f * (qx * qy + qz * qw), R11 = 1.f - 2.f * (qx * qx + qz * qz), R12 = 2.f * (qy * qz - qx * qw);
        float R20 = 2.f * (qx * qz - qy * qw), R21 = 2.f * (qy * qz + qx * qw), R22 = 1.f - 2.f * (qx * qx + qy * qy);
        float r0 = ct[0] - cp[0], r1 = ct[1] - cp[1], r2 = ct[2] - cp[2];
        // p_cam = R^T * rel
        float pc0 = R00 * r0 + R10 * r1 + R20 * r2;
        float pc1 = R01 * r0 + R11 * r1 + R21 * r2;
        float pc2 = R02 * r0 + R12 * r1 + R22 * r2;
        float zc = fmaxf(pc2, 0.01f);
        float u = it[2] * pc0 / zc + it[4];
        float v = it[3] * pc1 / zc + it[5];
        float p0 = v / it[0] * (float)NIc;   // continuous row in 64-grid
        float p1 = u / it[1] * (float)NJc;   // continuous col
        int px0 = (int)floorf(p0);
        int px1 = (int)floorf(p1);
        float off0 = fminf(fmaxf(p0 - 0.5f, 0.f), (float)(NIc - 1));
        float off1 = fminf(fmaxf(p1 - 0.5f, 0.f), (float)(NJc - 1));
        float wv[4]; int iv[4]; float wsum = 0.f;
#pragma unroll
        for (int s = 0; s < 4; ++s) {
            int s0 = s >> 1, s1 = s & 1;
            int q0 = iclamp(px0 + s0, 0, NIc - 1);
            int q1 = iclamp(px1 + s1, 0, NJc - 1);
            iv[s] = q0 * NJc + q1;
            float rr0 = fabsf(off0 - (float)(px0 + 1 - s0)) + 0.01f;
            float rr1 = fabsf(off1 - (float)(px1 + 1 - s1)) + 0.01f;
            wv[s] = rr0 * rr1;
            wsum += wv[s];
        }
        float inv = 1.f / wsum;
#pragma unroll
        for (int s = 0; s < 4; ++s) { s_w[c][s] = wv[s] * inv; s_idx[c][s] = iv[s]; }
    }
    __syncthreads();

    {   // weighted 4-tap gather: 64 threads per camera, float4 per thread
        int c = tid >> 6;
        int l = tid & 63;
        const float* base = img_feat + ((size_t)(b * NC + c) * (NIc * NJc)) * NF;
        float4 acc = make_float4(0.f, 0.f, 0.f, 0.f);
#pragma unroll
        for (int s = 0; s < 4; ++s) {
            const float4* p = (const float4*)(base + (size_t)s_idx[c][s] * NF + l * 4);
            float4 g = *p;
            float w = s_w[c][s];
            acc.x += w * g.x; acc.y += w * g.y; acc.z += w * g.z; acc.w += w * g.w;
        }
        s_feats[c][l * 4 + 0] = acc.x;
        s_feats[c][l * 4 + 1] = acc.y;
        s_feats[c][l * 4 + 2] = acc.z;
        s_feats[c][l * 4 + 3] = acc.w;
        __bf16* fd = feats_ws + ((size_t)row * NC + c) * NF + l * 4;
        fd[0] = (__bf16)acc.x; fd[1] = (__bf16)acc.y;
        fd[2] = (__bf16)acc.z; fd[3] = (__bf16)acc.w;
    }
    __syncthreads();

    if (tid < NC * NH) {            // attention logits
        int c = tid >> 3, h = tid & 7;
        float dot = b_attn[h];
        for (int f = 0; f < NF; ++f) dot += s_feats[c][f] * w_attn[f * NH + h];
        s_aw[c][h] = dot;
    }
    __syncthreads();
    if (tid < NC) {                 // softmax over heads
        int c = tid;
        float mx = s_aw[c][0];
#pragma unroll
        for (int h = 1; h < NH; ++h) mx = fmaxf(mx, s_aw[c][h]);
        float e[NH]; float sum = 0.f;
#pragma unroll
        for (int h = 0; h < NH; ++h) { e[h] = __expf(s_aw[c][h] - mx); sum += e[h]; }
        float inv = 1.f / sum;
#pragma unroll
        for (int h = 0; h < NH; ++h)
            aw_ws[((size_t)row * NC + c) * NH + h] = e[h] * inv;
    }
}

// -------------------------------------------------------------------------
// Kernel 2: fused dual GEMM via v_wmma_f32_16x16x32_bf16.
// Block = 16 object-rows, 256 threads = 8 waves.
// Stage 1 (wave w = head w): comb[16,128] = b_feat + sum_c aw*(feats_c@w_feat_h)
//   Loop order: kt outer (no unroll) -> build 4 aw-scaled A frags -> nt inner
//   loads each B frag ONCE and chains 4 WMMAs into acc[nt]. Nothing is
//   loop-invariant, so no LICM spill storm.
// Stage 2: out[16,256] = z + b_out + comb @ w_out, comb staged bf16 in LDS.
// -------------------------------------------------------------------------
__global__ __launch_bounds__(256, 1) void attn_gemm(
    const __bf16* __restrict__ feats_ws,    // (NB*NO*NC, NF)
    const float*  __restrict__ aw_ws,       // (NB*NO, NC, NH)
    const __bf16* __restrict__ wf_pk,       // packed w_feat frags
    const __bf16* __restrict__ wo_pk,       // packed w_out frags
    const float*  __restrict__ b_feat,      // (NH, D2)
    const float*  __restrict__ b_out,       // (ND)
    const float*  __restrict__ z,           // (NB,NO,1,ND)
    float* __restrict__ out)                // (NB,NO,1,ND)
{
    __shared__ __align__(16) __bf16 s_comb[16][1032];  // pad to dodge bank conflicts

    int lane = threadIdx.x & 31;
    int wv   = threadIdx.x >> 5;     // wave id == head id in stage 1
    int lh   = lane >> 4;            // lane half
    int nl   = lane & 15;            // N within tile / A row within tile
    int rowBase = blockIdx.x * 16;
    int m = nl;                      // this lane's A-matrix row

    // ---- Stage 1: per-head GEMM with aw-scaled A rows ----
    int h = wv;
    float awv[NC];
#pragma unroll
    for (int c = 0; c < NC; ++c)
        awv[c] = aw_ws[((size_t)(rowBase + m) * NC + c) * NH + h];

    v8f acc[8];
#pragma unroll
    for (int nt = 0; nt < 8; ++nt) {
        float bb = b_feat[h * D2 + nt * 16 + nl];
#pragma unroll
        for (int r = 0; r < 8; ++r) acc[nt][r] = bb;
    }

    const __bf16* arow0 = feats_ws + (size_t)(rowBase + m) * NC * NF; // cam 0 row
    const __bf16* wfbase = wf_pk + (size_t)h * 8 * 8 * 512 + lane * 16;

#pragma unroll 1
    for (int kt = 0; kt < 8; ++kt) {
        // Build all 4 aw-scaled A fragments for this kt (4 x 8 VGPRs).
        v16bf a[NC];
#pragma unroll
        for (int c = 0; c < NC; ++c) {
            const __bf16* ap = arow0 + c * NF + kt * 32 + 8 * lh;
            v8bf lo = *(const v8bf*)ap;          // global_load_b128
            v8bf hi = *(const v8bf*)(ap + 16);
            v16bf t = cat8(lo, hi);
            float s = awv[c];
#pragma unroll
            for (int i = 0; i < 16; ++i) t[i] = (__bf16)((float)t[i] * s);
            a[c] = t;
        }
        // Each B fragment loaded once, reused for 4 cameras.
#pragma unroll
        for (int nt = 0; nt < 8; ++nt) {
            v16bf bfr = *(const v16bf*)(wfbase + ((size_t)nt * 8 + kt) * 512);
#pragma unroll
            for (int c = 0; c < NC; ++c)
                acc[nt] = __builtin_amdgcn_wmma_f32_16x16x32_bf16(
                    false, a[c], false, bfr, (short)0, acc[nt], false, false);
        }
    }

    // comb -> LDS as bf16 (C/D layout: M = r + 8*lh, N = nl)
#pragma unroll
    for (int nt = 0; nt < 8; ++nt)
#pragma unroll
        for (int r = 0; r < 8; ++r)
            s_comb[r + 8 * lh][h * D2 + nt * 16 + nl] = (__bf16)acc[nt][r];
    __syncthreads();

    // ---- Stage 2: out = z + b_out + comb @ w_out ----
    v8f acc2[2];
#pragma unroll
    for (int t = 0; t < 2; ++t) {
        float bb = b_out[(wv * 2 + t) * 16 + nl];
#pragma unroll
        for (int r = 0; r < 8; ++r) acc2[t][r] = bb;
    }

    const __bf16* wobase = wo_pk + lane * 16;
#pragma unroll 2
    for (int kt = 0; kt < 32; ++kt) {
        const __bf16* ap = &s_comb[m][kt * 32 + 8 * lh];
        v8bf lo = *(const v8bf*)ap;              // ds_load_b128
        v8bf hi = *(const v8bf*)(ap + 16);
        v16bf a = cat8(lo, hi);
#pragma unroll
        for (int t = 0; t < 2; ++t) {
            int nt = wv * 2 + t;
            v16bf bfr = *(const v16bf*)(wobase + ((size_t)(nt * 32 + kt)) * 512);
            acc2[t] = __builtin_amdgcn_wmma_f32_16x16x32_bf16(
                false, a, false, bfr, (short)0, acc2[t], false, false);
        }
    }

#pragma unroll
    for (int t = 0; t < 2; ++t) {
        int nt = wv * 2 + t;
#pragma unroll
        for (int r = 0; r < 8; ++r) {
            int rr = r + 8 * lh;
            size_t idx = (size_t)(rowBase + rr) * ND + nt * 16 + nl;
            out[idx] = z[idx] + acc2[t][r];
        }
    }
}

// -------------------------------------------------------------------------
extern "C" void kernel_launch(void* const* d_in, const int* in_sizes, int n_in,
                              void* d_out, int out_size, void* d_ws, size_t ws_size,
                              hipStream_t stream) {
    const float* center      = (const float*)d_in[0];   // (16,1024,3)
    const float* cam_posquat = (const float*)d_in[1];   // (16,4,7)
    const float* intrinsic   = (const float*)d_in[2];   // (16,4,6)
    const float* img_feat    = (const float*)d_in[3];   // (16,4,64,64,256)
    const float* z           = (const float*)d_in[4];   // (16,1024,1,256)
    const float* w_attn      = (const float*)d_in[5];   // (256,8)
    const float* b_attn      = (const float*)d_in[6];   // (8)
    const float* w_feat      = (const float*)d_in[7];   // (256,8,128)
    const float* b_feat      = (const float*)d_in[8];   // (8,128)
    const float* w_out       = (const float*)d_in[9];   // (1024,256)
    const float* b_out       = (const float*)d_in[10];  // (256)

    char* ws = (char*)d_ws;
    // feats bf16: 16*1024*4*256*2 = 33,554,432 B
    __bf16* feats_ws = (__bf16*)(ws + 0);
    // aw f32: 16*1024*4*8*4 = 2,097,152 B
    float*  aw_ws    = (float*)(ws + 33554432);
    // packed w_feat: 512 frags * 512 bf16 = 524,288 B
    __bf16* wf_pk    = (__bf16*)(ws + 35651584);
    // packed w_out:  512 frags * 512 bf16 = 524,288 B
    __bf16* wo_pk    = (__bf16*)(ws + 36175872);

    pack_weights<<<128, 256, 0, stream>>>(w_feat, w_out, wf_pk, wo_pk);
    gather_feats<<<NB * NO, 256, 0, stream>>>(center, cam_posquat, intrinsic,
                                              img_feat, w_attn, b_attn,
                                              feats_ws, aw_ws);
    attn_gemm<<<(NB * NO) / 16, 256, 0, stream>>>(feats_ws, aw_ws, wf_pk, wo_pk,
                                                  b_feat, b_out, z, (float*)d_out);
}
